// Seq2Seq_31069793419877
// MI455X (gfx1250) — compile-verified, matrix operand order checked by reading, not run on hardware
//
#include <hip/hip_runtime.h>
#include <hip/hip_bf16.h>

typedef _Float16 half_t;
typedef __attribute__((ext_vector_type(16))) _Float16 v16h;
typedef __attribute__((ext_vector_type(8)))  float    v8f;

#if defined(__has_builtin)
#if __has_builtin(__builtin_amdgcn_global_load_async_to_lds_b128)
#define HAVE_ASYNC_B128 1
#endif
#if __has_builtin(__builtin_amdgcn_s_wait_asynccnt)
#define HAVE_WAIT_ASYNC 1
#endif
#endif

// async-to-LDS builtin operand types: int4 pointers in global (AS1) / LDS (AS3)
typedef int int4v __attribute__((ext_vector_type(4)));
typedef __attribute__((address_space(1))) int4v gint4;
typedef __attribute__((address_space(3))) int4v lint4;

#define B_  64
#define S_  64
#define T_  48
#define TD_ 47
#define E_  256
#define H_  512
#define V_  32000
#define G_  1536   // 3*H

__device__ __forceinline__ float sigmoid_(float x) { return 1.0f / (1.0f + expf(-x)); }

// ---------------------------------------------------------------------------
// Generic f16 GEMM: C[M,N] = A[M,K] @ W[N, wk0:wk0+K]^T (+bias[n]) (+addsrc[m,n])
// grid = (N/64, M/64), block = 256 threads (8 wave32).
// 64-wide K stages through LDS (async DMA to LDS when available); each wave
// computes one 16-row strip x two 16-col sub-tiles: 4 wmma per stage.
// ---------------------------------------------------------------------------
__global__ __launch_bounds__(256) void k_gemm_f16(
    const half_t* __restrict__ A, int lda,
    const half_t* __restrict__ W, int ldw, int wk0,
    float* __restrict__ C, long long ldc,
    int K,
    const float* __restrict__ bias,
    const float* __restrict__ addsrc, int ldadd)
{
    __shared__ __align__(16) half_t sA[64 * 64];
    __shared__ __align__(16) half_t sW[64 * 64];

    const int t      = threadIdx.x;
    const int tile_m = blockIdx.y * 64;
    const int tile_n = blockIdx.x * 64;
    const int wave   = t >> 5;
    const int lane   = t & 31;
    const int mi     = wave & 3;           // m sub-tile 0..3
    const int ni0    = (wave >> 2) * 2;    // n sub-tiles {0,1} or {2,3}

    const int lrow = t >> 2;               // 0..63 (staging row)
    const int lseg = (t & 3) << 4;         // 0,16,32,48 (staging k-segment, halfs)

    v8f acc0 = {};
    v8f acc1 = {};

    for (int k0 = 0; k0 < K; k0 += 64) {
        // cooperative stage: 64x64 halfs of A and W (2x16B per thread each)
        const half_t* gA = A + (size_t)(tile_m + lrow) * lda + (k0 + lseg);
        const half_t* gW = W + (size_t)(tile_n + lrow) * ldw + (wk0 + k0 + lseg);
        half_t* lA = &sA[lrow * 64 + lseg];
        half_t* lW = &sW[lrow * 64 + lseg];
#if HAVE_ASYNC_B128
        // CDNA5 async DMA: global -> LDS, no VGPR staging, tracked by ASYNCcnt
        __builtin_amdgcn_global_load_async_to_lds_b128(
            (gint4*)(gA),     (lint4*)(lA),     0, 0);
        __builtin_amdgcn_global_load_async_to_lds_b128(
            (gint4*)(gA + 8), (lint4*)(lA + 8), 0, 0);
        __builtin_amdgcn_global_load_async_to_lds_b128(
            (gint4*)(gW),     (lint4*)(lW),     0, 0);
        __builtin_amdgcn_global_load_async_to_lds_b128(
            (gint4*)(gW + 8), (lint4*)(lW + 8), 0, 0);
#if HAVE_WAIT_ASYNC
        __builtin_amdgcn_s_wait_asynccnt(0);
#else
        asm volatile("s_wait_asynccnt 0x0" ::: "memory");
#endif
#else
        *(uint4*)(lA)     = *(const uint4*)(gA);
        *(uint4*)(lA + 8) = *(const uint4*)(gA + 8);
        *(uint4*)(lW)     = *(const uint4*)(gW);
        *(uint4*)(lW + 8) = *(const uint4*)(gW + 8);
#endif
        __syncthreads();

        // fragment build per CDNA5 16-bit layouts:
        // lane<16: row=lane,   K = kg+0..7 (v0..3), 16+kg+0..7 (v4..7), kg=0
        // lane>=16: row=lane-16, same with kg=8
        const int hr = lane & 15;
        const int kg = (lane >> 4) << 3;
#pragma unroll
        for (int ks = 0; ks < 64; ks += 32) {
            const half_t* pa  = &sA[(mi  * 16 + hr) * 64 + ks + kg];
            const half_t* pb0 = &sW[(ni0 * 16 + hr) * 64 + ks + kg];
            const half_t* pb1 = &sW[((ni0 + 1) * 16 + hr) * 64 + ks + kg];
            v16h a, b0, b1;
#pragma unroll
            for (int j = 0; j < 8; ++j) {
                a[j]  = pa[j];   a[j + 8]  = pa[16 + j];
                b0[j] = pb0[j];  b0[j + 8] = pb0[16 + j];
                b1[j] = pb1[j];  b1[j + 8] = pb1[16 + j];
            }
            acc0 = __builtin_amdgcn_wmma_f32_16x16x32_f16(false, a, false, b0, (short)0, acc0, false, false);
            acc1 = __builtin_amdgcn_wmma_f32_16x16x32_f16(false, a, false, b1, (short)0, acc1, false, false);
        }
        __syncthreads();
    }

    // C/D layout: lane&15 = column, rows r + 8*(lane>=16)
    const int col0 = tile_n + ni0 * 16 + (lane & 15);
    const int col1 = col0 + 16;
    const int rowb = tile_m + mi * 16 + ((lane >> 4) << 3);
#pragma unroll
    for (int r = 0; r < 8; ++r) {
        const int m = rowb + r;
        float v0 = acc0[r], v1 = acc1[r];
        if (bias)   { v0 += bias[col0]; v1 += bias[col1]; }
        if (addsrc) { v0 += addsrc[(size_t)m * ldadd + col0];
                      v1 += addsrc[(size_t)m * ldadd + col1]; }
        C[(size_t)m * ldc + col0] = v0;
        C[(size_t)m * ldc + col1] = v1;
    }
}

// ---------------------------------------------------------------------------
// GRU nonlinearity (+ optional length mask, + optional encoder-output write)
// ---------------------------------------------------------------------------
__global__ void k_gru(const float* __restrict__ gi, const float* __restrict__ gh,
                      float* __restrict__ h, half_t* __restrict__ h16, int ld16,
                      const int* __restrict__ lens, int step,
                      float* __restrict__ eo32, half_t* __restrict__ eo16, int eostride)
{
    int idx = blockIdx.x * blockDim.x + threadIdx.x;
    if (idx >= B_ * H_) return;
    const int b = idx >> 9;
    const int j = idx & (H_ - 1);
    const float* gib = gi + (size_t)b * G_;
    const float* ghb = gh + (size_t)b * G_;
    const float hp = h[idx];
    const float r = sigmoid_(gib[j]        + ghb[j]);
    const float z = sigmoid_(gib[H_ + j]   + ghb[H_ + j]);
    const float n = tanhf   (gib[2*H_ + j] + r * ghb[2*H_ + j]);
    float hn = (1.0f - z) * n + z * hp;
    bool valid = true;
    if (lens) { valid = (step < lens[b]); if (!valid) hn = hp; }
    h[idx] = hn;
    h16[(size_t)b * ld16 + j] = (half_t)hn;
    if (eo32) {
        const float o = valid ? hn : 0.0f;
        eo32[(size_t)b * eostride + j] = o;
        eo16[(size_t)b * eostride + j] = (half_t)o;
    }
}

// ---------------------------------------------------------------------------
// Attention: one block per batch row. scores -> masked softmax -> ctx.
// Writes attn probs into d_out and ctx (f16) into cat[:,512:1024].
// ---------------------------------------------------------------------------
__global__ __launch_bounds__(256) void k_attn(
    const float* __restrict__ enc_proj, const float* __restrict__ decproj,
    const float* __restrict__ a_v, const int* __restrict__ src_ids,
    const float* __restrict__ enc_out,
    float* __restrict__ attn_out,       // base already offset by t*S_; stride per b = TD_*S_
    half_t* __restrict__ cat)
{
    __shared__ float s_sc[S_];
    __shared__ float s_at[S_];
    __shared__ float s_red[256];

    const int t = threadIdx.x;
    const int b = blockIdx.x;
    const int s = t >> 2;
    const int p = t & 3;

    // scores: 4 threads per source position, each reduces 128 of H
    const float* ep = enc_proj + ((size_t)b * S_ + s) * H_;
    const float* dp = decproj + (size_t)b * H_;
    float part = 0.0f;
    for (int hh = p * 128; hh < p * 128 + 128; ++hh)
        part += tanhf(ep[hh] + dp[hh]) * a_v[hh];
    s_red[t] = part;
    __syncthreads();
    if (p == 0) {
        float sc = s_red[t] + s_red[t + 1] + s_red[t + 2] + s_red[t + 3];
        s_sc[s] = (src_ids[b * S_ + s] != 0) ? sc : -1.0e9f;
    }
    __syncthreads();

    // softmax over 64 positions
    if (t < S_) s_red[t] = s_sc[t];
    __syncthreads();
    for (int off = 32; off >= 1; off >>= 1) {
        if (t < off) s_red[t] = fmaxf(s_red[t], s_red[t + off]);
        __syncthreads();
    }
    const float mx = s_red[0];
    __syncthreads();
    if (t < S_) { float e = expf(s_sc[t] - mx); s_at[t] = e; s_red[t] = e; }
    __syncthreads();
    for (int off = 32; off >= 1; off >>= 1) {
        if (t < off) s_red[t] += s_red[t + off];
        __syncthreads();
    }
    const float inv = 1.0f / s_red[0];
    __syncthreads();
    if (t < S_) {
        const float a = s_at[t] * inv;
        s_at[t] = a;
        attn_out[(size_t)b * (TD_ * S_) + t] = a;
    }
    __syncthreads();

    // ctx[b,h] = sum_s attn[s] * enc_out[b,s,h]  -> f16 into cat[:,512+h]
    for (int hh = t; hh < H_; hh += 256) {
        float c = 0.0f;
        for (int s2 = 0; s2 < S_; ++s2)
            c += s_at[s2] * enc_out[((size_t)b * S_ + s2) * H_ + hh];
        cat[(size_t)b * (2 * H_) + H_ + hh] = (half_t)c;
    }
}

// ---------------------------------------------------------------------------
// Utility kernels
// ---------------------------------------------------------------------------
__global__ void k_cvt(const float* __restrict__ src, half_t* __restrict__ dst, int n) {
    int i = blockIdx.x * blockDim.x + threadIdx.x;
    if (i < n) dst[i] = (half_t)src[i];
}
__global__ void k_zero_f32(float* p, int n) {
    int i = blockIdx.x * blockDim.x + threadIdx.x;
    if (i < n) p[i] = 0.0f;
}
__global__ void k_zero_f16(half_t* p, int n) {
    int i = blockIdx.x * blockDim.x + threadIdx.x;
    if (i < n) p[i] = (half_t)0.0f;
}
__global__ void k_copy_state(const float* __restrict__ src, float* __restrict__ dst,
                             half_t* __restrict__ dst16, int ld16) {
    int idx = blockIdx.x * blockDim.x + threadIdx.x;
    if (idx >= B_ * H_) return;
    const int b = idx >> 9, j = idx & (H_ - 1);
    const float v = src[idx];
    dst[idx] = v;
    dst16[(size_t)b * ld16 + j] = (half_t)v;
}
// X_enc[(s*B+b)*E + e] = src_emb[src_ids[b,s]] (f16)
__global__ void k_embed_enc(const int* __restrict__ ids, const float* __restrict__ emb,
                            half_t* __restrict__ X) {
    int idx = blockIdx.x * blockDim.x + threadIdx.x;
    if (idx >= S_ * B_ * E_) return;
    const int e = idx & (E_ - 1);
    const int sb = idx >> 8;
    const int b = sb & (B_ - 1);
    const int s = sb >> 6;
    const int id = ids[b * S_ + s];
    X[((size_t)(s * B_ + b)) * E_ + e] = (half_t)emb[(size_t)id * E_ + e];
}
// X_dec[(t*B+b)*E + e] = tgt_emb[tgt_ids[b,t]] for t in [0,46]
__global__ void k_embed_dec(const int* __restrict__ ids, const float* __restrict__ emb,
                            half_t* __restrict__ X) {
    int idx = blockIdx.x * blockDim.x + threadIdx.x;
    if (idx >= TD_ * B_ * E_) return;
    const int e = idx & (E_ - 1);
    const int tb = idx >> 8;
    const int b = tb & (B_ - 1);
    const int tt = tb >> 6;
    const int id = ids[b * T_ + tt];
    X[((size_t)(tt * B_ + b)) * E_ + e] = (half_t)emb[(size_t)id * E_ + e];
}

// ---------------------------------------------------------------------------
extern "C" void kernel_launch(void* const* d_in, const int* in_sizes, int n_in,
                              void* d_out, int out_size, void* d_ws, size_t ws_size,
                              hipStream_t stream) {
    (void)in_sizes; (void)n_in; (void)out_size; (void)ws_size;

    const int*   src_ids  = (const int*)d_in[0];
    const int*   src_lens = (const int*)d_in[1];
    const int*   tgt_ids  = (const int*)d_in[2];
    const float* src_emb  = (const float*)d_in[3];
    const float* tgt_emb  = (const float*)d_in[4];
    const float* eW_ih0 = (const float*)d_in[5];
    const float* eW_hh0 = (const float*)d_in[6];
    const float* eb_ih0 = (const float*)d_in[7];
    const float* eb_hh0 = (const float*)d_in[8];
    const float* eW_ih1 = (const float*)d_in[9];
    const float* eW_hh1 = (const float*)d_in[10];
    const float* eb_ih1 = (const float*)d_in[11];
    const float* eb_hh1 = (const float*)d_in[12];
    const float* dW_ih0 = (const float*)d_in[13];
    const float* dW_hh0 = (const float*)d_in[14];
    const float* db_ih0 = (const float*)d_in[15];
    const float* db_hh0 = (const float*)d_in[16];
    const float* dW_ih1 = (const float*)d_in[17];
    const float* dW_hh1 = (const float*)d_in[18];
    const float* db_ih1 = (const float*)d_in[19];
    const float* db_hh1 = (const float*)d_in[20];
    const float* aW_enc = (const float*)d_in[21];
    const float* aW_dec = (const float*)d_in[22];
    const float* a_v    = (const float*)d_in[23];
    const float* oW     = (const float*)d_in[24];
    const float* ob     = (const float*)d_in[25];

    float* out_logits = (float*)d_out;                                // (B, 47, V)
    float* out_attn   = out_logits + (size_t)B_ * TD_ * V_;           // (B, 47, S)

    // ---- workspace carve (256B aligned) ----
    char* ws = (char*)d_ws;
    size_t off = 0;
    auto alloc = [&](size_t bytes) -> void* {
        void* p = ws + off;
        off = (off + bytes + 255) & ~(size_t)255;
        return p;
    };
    half_t* eW_ih0_h = (half_t*)alloc(sizeof(half_t) * (size_t)G_ * E_);
    half_t* eW_hh0_h = (half_t*)alloc(sizeof(half_t) * (size_t)G_ * H_);
    half_t* eW_ih1_h = (half_t*)alloc(sizeof(half_t) * (size_t)G_ * H_);
    half_t* eW_hh1_h = (half_t*)alloc(sizeof(half_t) * (size_t)G_ * H_);
    half_t* dW_ih0_h = (half_t*)alloc(sizeof(half_t) * (size_t)G_ * (E_ + H_));
    half_t* dW_hh0_h = (half_t*)alloc(sizeof(half_t) * (size_t)G_ * H_);
    half_t* dW_ih1_h = (half_t*)alloc(sizeof(half_t) * (size_t)G_ * H_);
    half_t* dW_hh1_h = (half_t*)alloc(sizeof(half_t) * (size_t)G_ * H_);
    half_t* aW_enc_h = (half_t*)alloc(sizeof(half_t) * (size_t)H_ * H_);
    half_t* aW_dec_h = (half_t*)alloc(sizeof(half_t) * (size_t)H_ * H_);
    half_t* oW_h     = (half_t*)alloc(sizeof(half_t) * (size_t)V_ * 2 * H_);
    half_t* Xe       = (half_t*)alloc(sizeof(half_t) * (size_t)S_ * B_ * E_);
    half_t* Xd       = (half_t*)alloc(sizeof(half_t) * (size_t)TD_ * B_ * E_);
    half_t* encout16 = (half_t*)alloc(sizeof(half_t) * (size_t)B_ * S_ * H_);
    half_t* h0e16    = (half_t*)alloc(sizeof(half_t) * (size_t)B_ * H_);
    half_t* h1e16    = (half_t*)alloc(sizeof(half_t) * (size_t)B_ * H_);
    half_t* h0d16    = (half_t*)alloc(sizeof(half_t) * (size_t)B_ * H_);
    half_t* cat      = (half_t*)alloc(sizeof(half_t) * (size_t)B_ * 2 * H_);
    float* Gi0e     = (float*)alloc(sizeof(float) * (size_t)S_ * B_ * G_);
    float* Gi0d     = (float*)alloc(sizeof(float) * (size_t)TD_ * B_ * G_);
    float* encout32 = (float*)alloc(sizeof(float) * (size_t)B_ * S_ * H_);
    float* encproj  = (float*)alloc(sizeof(float) * (size_t)B_ * S_ * H_);
    float* h0e32    = (float*)alloc(sizeof(float) * (size_t)B_ * H_);
    float* h1e32    = (float*)alloc(sizeof(float) * (size_t)B_ * H_);
    float* h0d32    = (float*)alloc(sizeof(float) * (size_t)B_ * H_);
    float* h1d32    = (float*)alloc(sizeof(float) * (size_t)B_ * H_);
    float* gi_buf   = (float*)alloc(sizeof(float) * (size_t)B_ * G_);
    float* gh_buf   = (float*)alloc(sizeof(float) * (size_t)B_ * G_);
    float* decproj  = (float*)alloc(sizeof(float) * (size_t)B_ * H_);

    auto cvt = [&](const float* s, half_t* d, int n) {
        k_cvt<<<(n + 255) / 256, 256, 0, stream>>>(s, d, n);
    };
    auto gemm = [&](const half_t* A, int lda, const half_t* W, int ldw, int wk0,
                    float* C, long long ldc, int M, int N, int K,
                    const float* bias, const float* addsrc, int ldadd) {
        dim3 g(N / 64, M / 64);
        k_gemm_f16<<<g, 256, 0, stream>>>(A, lda, W, ldw, wk0, C, ldc, K, bias, addsrc, ldadd);
    };

    // ---- weight conversion to f16 ----
    cvt(eW_ih0, eW_ih0_h, G_ * E_);
    cvt(eW_hh0, eW_hh0_h, G_ * H_);
    cvt(eW_ih1, eW_ih1_h, G_ * H_);
    cvt(eW_hh1, eW_hh1_h, G_ * H_);
    cvt(dW_ih0, dW_ih0_h, G_ * (E_ + H_));
    cvt(dW_hh0, dW_hh0_h, G_ * H_);
    cvt(dW_ih1, dW_ih1_h, G_ * H_);
    cvt(dW_hh1, dW_hh1_h, G_ * H_);
    cvt(aW_enc, aW_enc_h, H_ * H_);
    cvt(aW_dec, aW_dec_h, H_ * H_);
    cvt(oW, oW_h, V_ * 2 * H_);

    // ---- init states ----
    k_zero_f32<<<(B_ * H_ + 255) / 256, 256, 0, stream>>>(h0e32, B_ * H_);
    k_zero_f32<<<(B_ * H_ + 255) / 256, 256, 0, stream>>>(h1e32, B_ * H_);
    k_zero_f16<<<(B_ * H_ + 255) / 256, 256, 0, stream>>>(h0e16, B_ * H_);
    k_zero_f16<<<(B_ * H_ + 255) / 256, 256, 0, stream>>>(h1e16, B_ * H_);

    // ---- embeddings + time-batched input projections ----
    k_embed_enc<<<(S_ * B_ * E_ + 255) / 256, 256, 0, stream>>>(src_ids, src_emb, Xe);
    gemm(Xe, E_, eW_ih0_h, E_, 0, Gi0e, G_, S_ * B_, G_, E_, eb_ih0, nullptr, 0);
    k_embed_dec<<<(TD_ * B_ * E_ + 255) / 256, 256, 0, stream>>>(tgt_ids, tgt_emb, Xd);
    gemm(Xd, E_, dW_ih0_h, E_ + H_, 0, Gi0d, G_, TD_ * B_, G_, E_, db_ih0, nullptr, 0);

    // ---- encoder ----
    for (int s = 0; s < S_; ++s) {
        // layer 0
        gemm(h0e16, H_, eW_hh0_h, H_, 0, gh_buf, G_, B_, G_, H_, eb_hh0, nullptr, 0);
        k_gru<<<(B_ * H_ + 255) / 256, 256, 0, stream>>>(
            Gi0e + (size_t)s * B_ * G_, gh_buf, h0e32, h0e16, H_,
            src_lens, s, nullptr, nullptr, 0);
        // layer 1
        gemm(h0e16, H_, eW_ih1_h, H_, 0, gi_buf, G_, B_, G_, H_, eb_ih1, nullptr, 0);
        gemm(h1e16, H_, eW_hh1_h, H_, 0, gh_buf, G_, B_, G_, H_, eb_hh1, nullptr, 0);
        k_gru<<<(B_ * H_ + 255) / 256, 256, 0, stream>>>(
            gi_buf, gh_buf, h1e32, h1e16, H_,
            src_lens, s, encout32 + (size_t)s * H_, encout16 + (size_t)s * H_, S_ * H_);
    }

    // enc_proj = enc_out @ aW_enc^T (all b,s at once)
    gemm(encout16, H_, aW_enc_h, H_, 0, encproj, H_, B_ * S_, H_, H_, nullptr, nullptr, 0);

    // ---- decoder init: h = h_enc, ctx = 0 ----
    k_zero_f16<<<(B_ * 2 * H_ + 255) / 256, 256, 0, stream>>>(cat, B_ * 2 * H_);
    k_copy_state<<<(B_ * H_ + 255) / 256, 256, 0, stream>>>(h0e32, h0d32, h0d16, H_);
    k_copy_state<<<(B_ * H_ + 255) / 256, 256, 0, stream>>>(h1e32, h1d32, cat, 2 * H_);

    // ---- decoder ----
    for (int t = 0; t < TD_; ++t) {
        // layer 0: gi = precomputed x-part + ctx @ dW_ih0[:,256:768]^T
        gemm(cat + H_, 2 * H_, dW_ih0_h, E_ + H_, E_, gi_buf, G_, B_, G_, H_,
             nullptr, Gi0d + (size_t)t * B_ * G_, G_);
        gemm(h0d16, H_, dW_hh0_h, H_, 0, gh_buf, G_, B_, G_, H_, db_hh0, nullptr, 0);
        k_gru<<<(B_ * H_ + 255) / 256, 256, 0, stream>>>(
            gi_buf, gh_buf, h0d32, h0d16, H_, nullptr, 0, nullptr, nullptr, 0);
        // layer 1 (h1 kept as f16 in cat[:, :512])
        gemm(h0d16, H_, dW_ih1_h, H_, 0, gi_buf, G_, B_, G_, H_, db_ih1, nullptr, 0);
        gemm(cat, 2 * H_, dW_hh1_h, H_, 0, gh_buf, G_, B_, G_, H_, db_hh1, nullptr, 0);
        k_gru<<<(B_ * H_ + 255) / 256, 256, 0, stream>>>(
            gi_buf, gh_buf, h1d32, cat, 2 * H_, nullptr, 0, nullptr, nullptr, 0);
        // attention
        gemm(cat, 2 * H_, aW_dec_h, H_, 0, decproj, H_, B_, H_, H_, nullptr, nullptr, 0);
        k_attn<<<B_, 256, 0, stream>>>(encproj, decproj, a_v, src_ids, encout32,
                                       out_attn + (size_t)t * S_, cat);
        // logits: [h1, ctx] @ oW^T + ob  -> d_out[b, t, :]
        gemm(cat, 2 * H_, oW_h, 2 * H_, 0, out_logits + (size_t)t * V_,
             (long long)TD_ * V_, B_, V_, 2 * H_, ob, nullptr, 0);
    }
}
